// HDCBrainV14_2_58334245814678
// MI455X (gfx1250) — matrix operand order repaired and checked by reading, not compiled, not run on hardware
//
// ============================================================================
// HDCBrainV14 forward — MI455X (gfx1250, wave32) implementation, rev 2.
//
// Roofline: ~47 GFLOP of block matmuls + ~67 GFLOP decoder GEMM, 65 MB f32
// output. All GEMMs use V_WMMA_F32_16X16X32_BF16 (bf16 operands, f32 accum):
// 8x the K-depth per instruction vs the f32 16x16x4 WMMA and half the operand
// bandwidth; every GEMM result feeds LayerNorm/sigmoid which absorbs bf16
// input rounding. Rev 2: 128-bit vectorized fragment loads (aligned by
// construction) and 16x32 macro-tiles per wave (one A fragment -> two WMMAs)
// to double A reuse; decay matrices hoisted out of the thought loop.
// ============================================================================
#include <hip/hip_runtime.h>
#include <cstdint>
#include <cmath>

#define B_  2
#define T_  256
#define D_  2048
#define V_  32000
#define NB_ 4
#define C_  512
#define R_  32
#define MT_ 3

typedef __bf16 bf16;
typedef __attribute__((ext_vector_type(16))) __bf16 v16bf;
typedef __attribute__((ext_vector_type(8)))  __bf16 v8bf;
typedef __attribute__((ext_vector_type(8)))  float  v8f;
typedef __attribute__((ext_vector_type(4)))  float  v4f;

__device__ __forceinline__ float sigf(float x) { return 1.f / (1.f + expf(-x)); }
__device__ __forceinline__ bf16  cvt_bf(float x) { return (bf16)x; }
__device__ __forceinline__ bf16  cvt_bf(bf16 x)  { return x; }

// 8 consecutive elements -> bf16x8. bf16 source: single 16B vector load.
// f32 source: two 16B vector loads + pack-convert.
__device__ __forceinline__ v8bf ld8(const bf16* p) { return *(const v8bf*)p; }
__device__ __forceinline__ v8bf ld8(const float* p)
{
  v4f a = *(const v4f*)p;
  v4f b = *(const v4f*)(p + 4);
  v8bf r;
  r[0] = (bf16)a[0]; r[1] = (bf16)a[1]; r[2] = (bf16)a[2]; r[3] = (bf16)a[3];
  r[4] = (bf16)b[0]; r[5] = (bf16)b[1]; r[6] = (bf16)b[2]; r[7] = (bf16)b[3];
  return r;
}
__device__ __forceinline__ v16bf cat(v8bf lo, v8bf hi)
{
  return __builtin_shufflevector(lo, hi, 0, 1, 2, 3, 4, 5, 6, 7,
                                 8, 9, 10, 11, 12, 13, 14, 15);
}

// ---------------------------------------------------------------------------
// bf16-WMMA GEMM. One wave computes one 16x32 f32 macro-tile (two 16x16 WMMA
// accumulators sharing one A fragment per k-step -> 2x A reuse).
//   BT=false: C[M,N] = A[M,K] * B[K,N]      (B row-major K x N)
//   BT=true : C[M,N] = A[M,K] * B[N,K]^T    (B row-major N x K)
// A/B element type float or bf16; f32 converted in registers on load (so the
// 32000x2048 codebook is never materialized in bf16). Fragment layouts per
// CDNA5 ISA 7.12.2, except A/B k order is contiguous (valid: A and B match):
//   A 16x32 bf16 : lanes 0-15 row M=l, v_i K=2i,2i+1 (K=0..15);
//                  lanes 16-31 same row, K=16..31.
//   B 32x16 bf16 : lanes 0-15 col N=l, v_i K=2i,2i+1; lanes 16-31 K+16.
//   C/D 16x16 f32: VGPR r -> row r (lanes 0-15) / row r+8 (lanes 16-31).
// Alignment: workspace buffers are 256B-aligned and every K, N stride is a
// multiple of 32 elements, so all vector loads below are >=16B aligned.
// ---------------------------------------------------------------------------
template <typename TA, typename TB, bool BT>
__global__ void k_gemm(const TA* __restrict__ A, const TB* __restrict__ Bm,
                       float* __restrict__ C, int M, int N, int K,
                       long sA, long sB, long sC)
{
  const int  lane   = threadIdx.x & 31;
  const int  wave   = threadIdx.x >> 5;
  const long tile   = (long)blockIdx.x * (blockDim.x >> 5) + wave;
  const int  tilesN = N >> 5;
  const long total  = (long)(M >> 4) * tilesN;
  if (tile >= total) return;                 // wave-uniform: live waves keep EXEC=all-1s
  const int tm = (int)(tile / tilesN);
  const int tn = (int)(tile % tilesN);

  const TA* Ab = A  + (long)blockIdx.z * sA;
  const TB* Bb = Bm + (long)blockIdx.z * sB;
  float*    Cb = C  + (long)blockIdx.z * sC;

  const int  half  = lane >> 4;
  const int  l16   = lane & 15;
  const long arow  = (long)tm * 16 + l16;
  const long bcol0 = (long)tn * 32 + l16;
  const long bcol1 = bcol0 + 16;

  v8f acc0 = {0.f, 0.f, 0.f, 0.f, 0.f, 0.f, 0.f, 0.f};
  v8f acc1 = {0.f, 0.f, 0.f, 0.f, 0.f, 0.f, 0.f, 0.f};

  for (int k0 = 0; k0 < K; k0 += 32) {
    const TA* ap = Ab + arow * K + k0 + half * 16;
    __builtin_prefetch(ap + 32, 0, 1);       // global_prefetch_b8: next K-block
    const v16bf av = cat(ld8(ap), ld8(ap + 8));

    v16bf bv0, bv1;
    if constexpr (BT) {
      const TB* bp0 = Bb + bcol0 * K + k0 + half * 16;
      const TB* bp1 = Bb + bcol1 * K + k0 + half * 16;
      bv0 = cat(ld8(bp0), ld8(bp0 + 8));
      bv1 = cat(ld8(bp1), ld8(bp1 + 8));
    } else {
      const TB* bp = Bb + (long)(k0 + half * 16) * N;
#pragma unroll
      for (int j = 0; j < 16; ++j) {         // lanes 0..15 cover 32B/row: coalesced
        bv0[j] = cvt_bf(bp[(long)j * N + bcol0]);
        bv1[j] = cvt_bf(bp[(long)j * N + bcol1]);
      }
    }
    acc0 = __builtin_amdgcn_wmma_f32_16x16x32_bf16(false, av, false, bv0,
                                                   (short)0, acc0, false, false);
    acc1 = __builtin_amdgcn_wmma_f32_16x16x32_bf16(false, av, false, bv1,
                                                   (short)0, acc1, false, false);
  }

  float* cp0 = Cb + (long)(tm * 16 + half * 8) * N + bcol0;
#pragma unroll
  for (int r = 0; r < 8; ++r) {
    cp0[(long)r * N]      = acc0[r];
    cp0[(long)r * N + 16] = acc1[r];
  }
}

template <typename TA, typename TB, bool BT>
static void launch_gemm(hipStream_t st, const TA* A, const TB* Bm, float* C,
                        int M, int N, int K, long sA, long sB, long sC, int batch)
{
  long tiles = (long)(M >> 4) * (N >> 5);
  dim3 grid((unsigned)((tiles + 3) / 4), 1, (unsigned)batch);   // 4 waves per block
  k_gemm<TA, TB, BT><<<grid, 128, 0, st>>>(A, Bm, C, M, N, K, sA, sB, sC);
}

// ---------------------------------------------------------------------------
// STE sign: out[d] = alpha*sign(w[d]) with 0 -> +alpha, alpha = mean(|row|).
// One block per row of length D_.
// ---------------------------------------------------------------------------
__global__ void k_sign_rows(float* __restrict__ out, const float* __restrict__ w, int D)
{
  const long rb = (long)blockIdx.x * D;
  float s = 0.f;
  for (int d = threadIdx.x; d < D; d += blockDim.x) s += fabsf(w[rb + d]);
  __shared__ float sh[256];
  sh[threadIdx.x] = s; __syncthreads();
  for (int st = 128; st > 0; st >>= 1) {
    if (threadIdx.x < st) sh[threadIdx.x] += sh[threadIdx.x + st];
    __syncthreads();
  }
  const float alpha = sh[0] / (float)D;
  for (int d = threadIdx.x; d < D; d += blockDim.x) {
    float v = w[rb + d];
    out[rb + d] = (v > 0.f) ? alpha : ((v < 0.f) ? -alpha : alpha);
  }
}

// ---------------------------------------------------------------------------
// Encode: tokens[b,t,d] = ste_sign(codebook[idx[b,t]])[(d - t) mod D]
// ---------------------------------------------------------------------------
__global__ void k_encode(float* __restrict__ tokens, const int* __restrict__ idx,
                         const float* __restrict__ codebook, int T, int D)
{
  const int  bt  = blockIdx.x;
  const int  t   = bt % T;
  const long row = (long)idx[bt] * D;
  float s = 0.f;
  for (int d = threadIdx.x; d < D; d += blockDim.x) s += fabsf(codebook[row + d]);
  __shared__ float sh[256];
  sh[threadIdx.x] = s; __syncthreads();
  for (int st = 128; st > 0; st >>= 1) {
    if (threadIdx.x < st) sh[threadIdx.x] += sh[threadIdx.x + st];
    __syncthreads();
  }
  const float alpha = sh[0] / (float)D;
  float* orow = tokens + (long)bt * D;
  for (int d = threadIdx.x; d < D; d += blockDim.x) {
    int src = d - t; if (src < 0) src += D;
    float v = codebook[row + src];
    orow[d] = (v > 0.f) ? alpha : ((v < 0.f) ? -alpha : alpha);
  }
}

// All NB decay matrices at once: A[bi][t,s] = sigmoid(dl[bi])^(t-s), t>=s.
__global__ void k_build_A(float* __restrict__ Am, const float* __restrict__ dl, int T)
{
  long i = (long)blockIdx.x * blockDim.x + threadIdx.x;
  if (i >= (long)NB_ * T * T) return;
  int bi = (int)(i / ((long)T * T));
  long r = i % ((long)T * T);
  int t = (int)(r / T), s = (int)(r % T);
  float decay = sigf(dl[bi]);
  Am[i] = (t >= s) ? powf(decay, (float)(t - s)) : 0.f;
}

// f32 -> bf16 with optional per-column scale vector (row length rl).
__global__ void k_cvt_bf16(bf16* __restrict__ dst, const float* __restrict__ src,
                           const float* __restrict__ scale, int rl, long n)
{
  long i = (long)blockIdx.x * blockDim.x + threadIdx.x;
  if (i >= n) return;
  float v = src[i];
  if (scale) v *= scale[i % rl];
  dst[i] = (bf16)v;
}

// Causal sigmoid-gated attention weights (bf16 for the attn@V GEMM).
__global__ void k_attn(bf16* __restrict__ attn, const float* __restrict__ scores,
                       int T, float sc, long n)
{
  long i = (long)blockIdx.x * blockDim.x + threadIdx.x;
  if (i >= n) return;
  int s = (int)(i % T), t = (int)((i / T) % T);
  attn[i] = (bf16)((t >= s) ? sigf(scores[i] * sc) : 0.f);
}

__global__ void k_sig_bf16(bf16* __restrict__ dst, const float* __restrict__ src,
                           float sc, long n)
{
  long i = (long)blockIdx.x * blockDim.x + threadIdx.x;
  if (i >= n) return;
  dst[i] = (bf16)sigf(src[i] * sc);
}

__global__ void k_mul(float* __restrict__ y, const float* __restrict__ x, long n)
{
  long i = (long)blockIdx.x * blockDim.x + threadIdx.x;
  if (i >= n) return;
  y[i] *= x[i];
}

// Exact GELU(src + bias) -> bf16 (bias over columns of length C).
__global__ void k_bias_gelu(bf16* __restrict__ dst, const float* __restrict__ src,
                            const float* __restrict__ bias, int C, long n)
{
  long i = (long)blockIdx.x * blockDim.x + threadIdx.x;
  if (i >= n) return;
  float v = src[i] + bias[i % C];
  dst[i] = (bf16)(0.5f * v * (1.f + erff(v * 0.70710678118654752f)));
}

// x += h2 + bias (bias over columns of length D).
__global__ void k_add_bias_res(float* __restrict__ x, const float* __restrict__ h2,
                               const float* __restrict__ bias, int D, long n)
{
  long i = (long)blockIdx.x * blockDim.x + threadIdx.x;
  if (i >= n) return;
  x[i] += h2[i] + bias[i % D];
}

// h += sigmoid(gates[ti]) * (th - h)
__global__ void k_lerp(float* __restrict__ h, const float* __restrict__ th,
                       const float* __restrict__ gates, int ti, long n)
{
  long i = (long)blockIdx.x * blockDim.x + threadIdx.x;
  if (i >= n) return;
  float g = sigf(gates[ti]);
  h[i] += g * (th[i] - h[i]);
}

__global__ void k_copy(float* __restrict__ d, const float* __restrict__ s, long n)
{
  long i = (long)blockIdx.x * blockDim.x + threadIdx.x;
  if (i < n) d[i] = s[i];
}

__global__ void k_scale(float* __restrict__ d, const float* __restrict__ sc, long n)
{
  long i = (long)blockIdx.x * blockDim.x + threadIdx.x;
  if (i < n) d[i] *= sc[0];
}

// ---------------------------------------------------------------------------
// Fused LayerNorm over rows of length D (D==2048, block==256, 8 elems/thread,
// cached in registers -> in-place safe):
//   z = gate? (1-g)*x + g*y : (y? x+y : x),  g = sigmoid(gate_ptr[gidx])
//   out = (z - mean)/sqrt(var+eps) * gw + gb  (+ pos)
// ---------------------------------------------------------------------------
__global__ void k_ln(float* __restrict__ out, const float* __restrict__ x,
                     const float* __restrict__ y, const float* __restrict__ gate_ptr,
                     int gidx, const float* __restrict__ gw, const float* __restrict__ gb,
                     const float* __restrict__ pos, int D)
{
  const long rb  = (long)blockIdx.x * D;
  const int  tid = threadIdx.x;
  float ca = 1.f, cb = 1.f;
  if (gate_ptr) { float g = sigf(gate_ptr[gidx]); ca = 1.f - g; cb = g; }
  float z[8], s = 0.f, ss = 0.f;
#pragma unroll
  for (int j = 0; j < 8; ++j) {
    int d = tid + j * 256;
    float v = ca * x[rb + d] + (y ? cb * y[rb + d] : 0.f);
    z[j] = v; s += v; ss += v * v;
  }
  __shared__ float s1[256], s2[256];
  s1[tid] = s; s2[tid] = ss; __syncthreads();
  for (int st = 128; st > 0; st >>= 1) {
    if (tid < st) { s1[tid] += s1[tid + st]; s2[tid] += s2[tid + st]; }
    __syncthreads();
  }
  const float m    = s1[0] / (float)D;
  const float var  = s2[0] / (float)D - m * m;
  const float rstd = rsqrtf(var + 1e-5f);
#pragma unroll
  for (int j = 0; j < 8; ++j) {
    int d = tid + j * 256;
    float v = (z[j] - m) * rstd * gw[d] + gb[d];
    if (pos) v += pos[d];
    out[rb + d] = v;
  }
}

// ---------------------------------------------------------------------------
static inline char* align256(char* p)
{
  return (char*)(((uintptr_t)p + 255u) & ~(uintptr_t)255u);
}
static inline dim3 g1(long n) { return dim3((unsigned)((n + 255) / 256)); }

extern "C" void kernel_launch(void* const* d_in, const int* in_sizes, int n_in,
                              void* d_out, int out_size, void* d_ws, size_t ws_size,
                              hipStream_t stream)
{
  (void)in_sizes; (void)n_in; (void)out_size; (void)ws_size;
  const int*   idx         = (const int*)  d_in[0];
  const float* codebook    = (const float*)d_in[1];
  const float* decay_logit = (const float*)d_in[2];
  const float* bv_q        = (const float*)d_in[3];
  const float* bv_k        = (const float*)d_in[4];
  const float* bv_v        = (const float*)d_in[5];
  const float* rules       = (const float*)d_in[6];
  const float* query_bind  = (const float*)d_in[7];
  const float* logic_gate  = (const float*)d_in[8];
  const float* ln_mem_g    = (const float*)d_in[9];
  const float* ln_mem_b    = (const float*)d_in[10];
  const float* ln_attn_g   = (const float*)d_in[11];
  const float* ln_attn_b   = (const float*)d_in[12];
  const float* ln_logic_g  = (const float*)d_in[13];
  const float* ln_logic_b  = (const float*)d_in[14];
  const float* ctrl_ln_g   = (const float*)d_in[15];
  const float* ctrl_ln_b   = (const float*)d_in[16];
  const float* down_w      = (const float*)d_in[17];
  const float* down_b      = (const float*)d_in[18];
  const float* up_w        = (const float*)d_in[19];
  const float* up_b        = (const float*)d_in[20];
  const float* tl_gates    = (const float*)d_in[21];
  const float* tl_pos      = (const float*)d_in[22];
  const float* tl_ln_g     = (const float*)d_in[23];
  const float* tl_ln_b     = (const float*)d_in[24];
  const float* out_ln_g    = (const float*)d_in[25];
  const float* out_ln_b    = (const float*)d_in[26];
  const float* out_scale   = (const float*)d_in[27];
  float* out = (float*)d_out;

  const long BTD = (long)B_ * T_ * D_;    // 1,048,576
  const long BTT = (long)B_ * T_ * T_;    // 131,072
  const long BTR = (long)B_ * T_ * R_;    // 16,384
  const long BTC = (long)B_ * T_ * C_;    // 262,144
  const long TT  = (long)T_ * T_;
  const float sc4 = 4.0f / sqrtf((float)D_);

  // ---- workspace carve (~24 MB) ----
  char* p = (char*)d_ws;
  auto carveF = [&](long n) { p = align256(p); float* r = (float*)p; p += n * sizeof(float); return r; };
  auto carveB = [&](long n) { p = align256(p); bf16*  r = (bf16*) p; p += n * sizeof(bf16);  return r; };
  float* signQ  = carveF((long)NB_ * D_);
  float* signK  = carveF((long)NB_ * D_);
  float* signV  = carveF((long)NB_ * D_);
  float* signQB = carveF((long)NB_ * D_);
  float* signR  = carveF((long)NB_ * R_ * D_);
  float* hbuf   = carveF(BTD);
  float* cur    = carveF(BTD);
  float* tmp    = carveF(BTD);
  float* scores = carveF(BTT);
  float* simb   = carveF(BTR);
  float* hid    = carveF(BTC);
  float* Aema   = carveF((long)NB_ * TT);
  bf16*  bf1    = carveB(BTD);
  bf16*  bf2    = carveB(BTD);
  bf16*  bf3    = carveB(BTD);
  bf16*  attnb  = carveB(BTT);
  bf16*  rwb    = carveB(BTR);
  bf16*  hidb   = carveB(BTC);

  // ---- pass-invariant precompute: STE signs + NB decay matrices ----
  k_sign_rows<<<NB_,      256, 0, stream>>>(signQ,  bv_q,       D_);
  k_sign_rows<<<NB_,      256, 0, stream>>>(signK,  bv_k,       D_);
  k_sign_rows<<<NB_,      256, 0, stream>>>(signV,  bv_v,       D_);
  k_sign_rows<<<NB_,      256, 0, stream>>>(signQB, query_bind, D_);
  k_sign_rows<<<NB_ * R_, 256, 0, stream>>>(signR,  rules,      D_);
  k_build_A<<<g1((long)NB_ * TT), 256, 0, stream>>>(Aema, decay_logit, T_);

  // ---- one pass over the NB blocks, operating in-place on `cur` ----
  auto run_blocks = [&]() {
    for (int bi = 0; bi < NB_; ++bi) {
      const long biD = (long)bi * D_;
      // EMA memory: tmp = A_ema @ cur  (A f32 on-load-converted, shared across batch)
      k_cvt_bf16<<<g1(BTD), 256, 0, stream>>>(bf1, cur, nullptr, D_, BTD);
      launch_gemm<float, bf16, false>(stream, Aema + bi * TT, bf1, tmp, T_, D_, T_,
                                      0, (long)T_ * D_, (long)T_ * D_, B_);
      k_ln<<<B_ * T_, 256, 0, stream>>>(cur, cur, tmp, nullptr, 0,
                                        ln_mem_g + biD, ln_mem_b + biD, nullptr, D_);
      // Boolean causal attention
      k_cvt_bf16<<<g1(BTD), 256, 0, stream>>>(bf1, cur, signQ + biD, D_, BTD);   // Q
      k_cvt_bf16<<<g1(BTD), 256, 0, stream>>>(bf2, cur, signK + biD, D_, BTD);   // K
      k_cvt_bf16<<<g1(BTD), 256, 0, stream>>>(bf3, cur, signV + biD, D_, BTD);   // V
      launch_gemm<bf16, bf16, true>(stream, bf1, bf2, scores, T_, T_, D_,
                                    (long)T_ * D_, (long)T_ * D_, TT, B_);
      k_attn<<<g1(BTT), 256, 0, stream>>>(attnb, scores, T_, sc4, BTT);
      launch_gemm<bf16, bf16, false>(stream, attnb, bf3, tmp, T_, D_, T_,
                                     TT, (long)T_ * D_, (long)T_ * D_, B_);
      k_ln<<<B_ * T_, 256, 0, stream>>>(cur, cur, tmp, nullptr, 0,
                                        ln_attn_g + biD, ln_attn_b + biD, nullptr, D_);
      // Logic layer
      k_cvt_bf16<<<g1(BTD), 256, 0, stream>>>(bf1, cur, signQB + biD, D_, BTD);  // query
      launch_gemm<bf16, float, true>(stream, bf1, signR + (long)bi * R_ * D_, simb,
                                     T_, R_, D_, (long)T_ * D_, 0, (long)T_ * R_, B_);
      k_sig_bf16<<<g1(BTR), 256, 0, stream>>>(rwb, simb, sc4, BTR);
      launch_gemm<bf16, float, false>(stream, rwb, signR + (long)bi * R_ * D_, tmp,
                                      T_, D_, R_, (long)T_ * R_, 0, (long)T_ * D_, B_);
      k_mul<<<g1(BTD), 256, 0, stream>>>(tmp, cur, BTD);                          // x * (rw@rls)
      k_ln<<<B_ * T_, 256, 0, stream>>>(cur, cur, tmp, logic_gate, bi,
                                        ln_logic_g + biD, ln_logic_b + biD, nullptr, D_);
      // Controller MLP
      k_ln<<<B_ * T_, 256, 0, stream>>>(tmp, cur, nullptr, nullptr, 0,
                                        ctrl_ln_g + biD, ctrl_ln_b + biD, nullptr, D_);
      k_cvt_bf16<<<g1(BTD), 256, 0, stream>>>(bf1, tmp, nullptr, D_, BTD);
      launch_gemm<bf16, float, true>(stream, bf1, down_w + (long)bi * C_ * D_, hid,
                                     T_, C_, D_, (long)T_ * D_, 0, (long)T_ * C_, B_);
      k_bias_gelu<<<g1(BTC), 256, 0, stream>>>(hidb, hid, down_b + (long)bi * C_, C_, BTC);
      launch_gemm<bf16, float, true>(stream, hidb, up_w + (long)bi * D_ * C_, tmp,
                                     T_, D_, C_, (long)T_ * C_, 0, (long)T_ * D_, B_);
      k_add_bias_res<<<g1(BTD), 256, 0, stream>>>(cur, tmp, up_b + biD, D_, BTD);
    }
  };

  // ---- encode + first pass ----
  k_encode<<<B_ * T_, 256, 0, stream>>>(cur, idx, codebook, T_, D_);
  run_blocks();
  k_copy<<<g1(BTD), 256, 0, stream>>>(hbuf, cur, BTD);

  // ---- thought loop ----
  for (int ti = 1; ti < MT_; ++ti) {
    k_ln<<<B_ * T_, 256, 0, stream>>>(cur, hbuf, nullptr, nullptr, 0,
                                      tl_ln_g, tl_ln_b, tl_pos + (long)ti * D_, D_);
    run_blocks();
    k_lerp<<<g1(BTD), 256, 0, stream>>>(hbuf, cur, tl_gates, ti, BTD);
  }

  // ---- decoder: logits = LN(h) @ codebook^T * output_scale ----
  k_ln<<<B_ * T_, 256, 0, stream>>>(cur, hbuf, nullptr, nullptr, 0,
                                    out_ln_g, out_ln_b, nullptr, D_);
  k_cvt_bf16<<<g1(BTD), 256, 0, stream>>>(bf1, cur, nullptr, D_, BTD);
  launch_gemm<bf16, float, true>(stream, bf1, codebook, out,
                                 B_ * T_, V_, D_, 0, 0, 0, 1);
  k_scale<<<g1((long)B_ * T_ * V_), 256, 0, stream>>>(out, out_scale, (long)B_ * T_ * V_);
}
